// StandardTransformerBlock_33363305955946
// MI455X (gfx1250) — matrix operand order repaired and checked by reading, not compiled, not run on hardware
//
#include <hip/hip_runtime.h>
#include <hip/hip_bf16.h>
#include <math.h>

#define ED   1024          // embed dim
#define HD   4096          // ffn hidden
#define NH   16            // heads
#define DHD  64            // head dim
#define NB   2             // batch
#define NSEQ 2048          // seq len
#define NTOK (NB*NSEQ)     // 4096 tokens

typedef __attribute__((ext_vector_type(16))) _Float16 v16h;
typedef __attribute__((ext_vector_type(8)))  _Float16 v8h;
typedef __attribute__((ext_vector_type(4)))  _Float16 v4h;
typedef __attribute__((ext_vector_type(8)))  float    v8f;

#if defined(__AMDGCN__) && \
    __has_builtin(__builtin_amdgcn_global_load_async_to_lds_b128) && \
    __has_builtin(__builtin_amdgcn_s_wait_asynccnt)
#define USE_ASYNC_LDS 1
typedef __attribute__((ext_vector_type(4))) int i4;
typedef __attribute__((address_space(1))) i4 gi4;   // global int4 (prints as "__device__")
typedef __attribute__((address_space(3))) i4 li4;   // LDS int4
#else
#define USE_ASYNC_LDS 0
#endif

union H16 { v16h v; v8h h[2]; };

__device__ __forceinline__ v16h load_frag(const _Float16* p0, const _Float16* p1) {
    H16 u;
    u.h[0] = *(const v8h*)p0;
    u.h[1] = *(const v8h*)p1;
    return u.v;
}

__device__ __forceinline__ v8f wmma16(v16h a, v16h b, v8f c) {
    // D = A(16x32 f16) * B(32x16 f16) + C(16x16 f32)
    return __builtin_amdgcn_wmma_f32_16x16x32_f16(false, a, false, b, (short)0, c, false, false);
}

__device__ __forceinline__ v8f vzero8() {
    v8f z = {0.f,0.f,0.f,0.f,0.f,0.f,0.f,0.f};
    return z;
}

// ---------------- f32 -> f16 conversion (weights) ----------------
__global__ __launch_bounds__(256) void cvt_kernel(const float* __restrict__ in,
                                                  _Float16* __restrict__ out, int n4) {
    int i = blockIdx.x * blockDim.x + threadIdx.x;
    if (i < n4) {
        float4 v = ((const float4*)in)[i];
        v4h h = {(_Float16)v.x, (_Float16)v.y, (_Float16)v.z, (_Float16)v.w};
        ((v4h*)out)[i] = h;
    }
}

// ---------------- LayerNorm: f32 in -> f16 out ----------------
__global__ __launch_bounds__(256) void ln_kernel(const float* __restrict__ x,
                                                 const float* __restrict__ g,
                                                 const float* __restrict__ bt,
                                                 _Float16* __restrict__ out) {
    __shared__ float red[256];
    const int row = blockIdx.x, t = threadIdx.x;
    const float* xr = x + (size_t)row * ED;
    float4 v = *(const float4*)(xr + t * 4);
    red[t] = v.x + v.y + v.z + v.w;
    __syncthreads();
    for (int off = 128; off; off >>= 1) { if (t < off) red[t] += red[t + off]; __syncthreads(); }
    float mean = red[0] * (1.0f / ED);
    __syncthreads();
    float d0 = v.x - mean, d1 = v.y - mean, d2 = v.z - mean, d3 = v.w - mean;
    red[t] = d0*d0 + d1*d1 + d2*d2 + d3*d3;
    __syncthreads();
    for (int off = 128; off; off >>= 1) { if (t < off) red[t] += red[t + off]; __syncthreads(); }
    float rstd = rsqrtf(red[0] * (1.0f / ED) + 1e-5f);
    float4 gg = *(const float4*)(g + t * 4);
    float4 bb = *(const float4*)(bt + t * 4);
    v4h o = {(_Float16)(d0 * rstd * gg.x + bb.x),
             (_Float16)(d1 * rstd * gg.y + bb.y),
             (_Float16)(d2 * rstd * gg.z + bb.z),
             (_Float16)(d3 * rstd * gg.w + bb.w)};
    *(v4h*)(out + (size_t)row * ED + t * 4) = o;
}

// ---------------- generic WMMA GEMM: C[M,N] = A[M,K] * W[N,K]^T ----------------
enum { M_QK = 0, M_VT = 1, M_RES = 2, M_GELU = 3, M_FINAL = 4 };

template <int MODE>
__global__ __launch_bounds__(256) void gemm_kernel(
        const _Float16* __restrict__ A, const _Float16* __restrict__ W,
        const float* __restrict__ bias, const float* __restrict__ res,
        float* __restrict__ outF, _Float16* __restrict__ outH,
        int Nn, int K, float qscale) {
    constexpr int BM = 128, BN = 128, BK = 32, LDT = 40;   // 40-half padded rows
    constexpr int TSZ = BM * LDT;                          // halfs per buffer
    __shared__ __align__(16) _Float16 sA[2 * TSZ];
    __shared__ __align__(16) _Float16 sW[2 * TSZ];

    const int tid  = threadIdx.x;
    const int lane = tid & 31, wave = tid >> 5;
    const int l15  = lane & 15, half = lane >> 4;
    const int m0   = blockIdx.y * BM, n0 = blockIdx.x * BN;
    const int wm   = (wave & 1) * 64;     // 2 waves over M: 64 rows each
    const int wn   = (wave >> 1) * 32;    // 4 waves over N: 32 cols each

    v8f acc[4][2];
#pragma unroll
    for (int i = 0; i < 4; i++)
#pragma unroll
        for (int j = 0; j < 2; j++) acc[i][j] = vzero8();

    const int ldr = tid >> 1;            // 0..127 (tile row)
    const int ldc = (tid & 1) * 16;      // 0 or 16 (k offset)
    const _Float16* Ar = A + (size_t)(m0 + ldr) * K + ldc;
    const _Float16* Wr = W + (size_t)(n0 + ldr) * K + ldc;
    _Float16* sAr = sA + ldr * LDT + ldc;
    _Float16* sWr = sW + ldr * LDT + ldc;

#if USE_ASYNC_LDS
    // Async global->LDS (ASYNCcnt), double-buffered: tile k+1 streams into LDS
    // while tile k feeds the WMMAs. 4 async b128 per thread per tile.
    {
        _Float16* a0 = const_cast<_Float16*>(Ar);
        _Float16* w0 = const_cast<_Float16*>(Wr);
        __builtin_amdgcn_global_load_async_to_lds_b128((gi4*)a0, (li4*)sAr, 0, 0);
        __builtin_amdgcn_global_load_async_to_lds_b128((gi4*)a0, (li4*)sAr, 16, 0);
        __builtin_amdgcn_global_load_async_to_lds_b128((gi4*)w0, (li4*)sWr, 0, 0);
        __builtin_amdgcn_global_load_async_to_lds_b128((gi4*)w0, (li4*)sWr, 16, 0);
    }
    int buf = 0;
    for (int k0 = 0; k0 < K; k0 += BK) {
        if (k0 + BK < K) {
            _Float16* an  = const_cast<_Float16*>(Ar + k0 + BK);
            _Float16* wn2 = const_cast<_Float16*>(Wr + k0 + BK);
            _Float16* sAn = sAr + (buf ^ 1) * TSZ;
            _Float16* sWn = sWr + (buf ^ 1) * TSZ;
            __builtin_amdgcn_global_load_async_to_lds_b128((gi4*)an,  (li4*)sAn, 0, 0);
            __builtin_amdgcn_global_load_async_to_lds_b128((gi4*)an,  (li4*)sAn, 16, 0);
            __builtin_amdgcn_global_load_async_to_lds_b128((gi4*)wn2, (li4*)sWn, 0, 0);
            __builtin_amdgcn_global_load_async_to_lds_b128((gi4*)wn2, (li4*)sWn, 16, 0);
            __builtin_amdgcn_s_wait_asynccnt(4);   // tile-k loads (issued earlier) done
        } else {
            __builtin_amdgcn_s_wait_asynccnt(0);
        }
        __syncthreads();                           // all waves' tile-k data visible

        const _Float16* sAb = sA + buf * TSZ;
        const _Float16* sWb = sW + buf * TSZ;
        v16h af[4], bf[2];
        const int aoff = half * 8;
#pragma unroll
        for (int i = 0; i < 4; i++) {
            const _Float16* p = sAb + (wm + i * 16 + l15) * LDT;
            af[i] = load_frag(p + aoff, p + aoff + 16);
        }
        const int boff = half * 8;
#pragma unroll
        for (int j = 0; j < 2; j++) {
            const _Float16* p = sWb + (wn + j * 16 + l15) * LDT;
            bf[j] = load_frag(p + boff, p + boff + 16);
        }
#pragma unroll
        for (int i = 0; i < 4; i++)
#pragma unroll
            for (int j = 0; j < 2; j++)
                acc[i][j] = wmma16(af[i], bf[j], acc[i][j]);
        __syncthreads();                           // buffer reuse guard
        buf ^= 1;
    }
#else
    // Fallback: VGPR-staged single buffer
    for (int k0 = 0; k0 < K; k0 += BK) {
        *(v8h*)(sAr)     = *(const v8h*)(Ar + k0);
        *(v8h*)(sAr + 8) = *(const v8h*)(Ar + k0 + 8);
        *(v8h*)(sWr)     = *(const v8h*)(Wr + k0);
        *(v8h*)(sWr + 8) = *(const v8h*)(Wr + k0 + 8);
        __syncthreads();
        v16h af[4], bf[2];
        const int aoff = half * 8;
#pragma unroll
        for (int i = 0; i < 4; i++) {
            const _Float16* p = sA + (wm + i * 16 + l15) * LDT;
            af[i] = load_frag(p + aoff, p + aoff + 16);
        }
        const int boff = half * 8;
#pragma unroll
        for (int j = 0; j < 2; j++) {
            const _Float16* p = sW + (wn + j * 16 + l15) * LDT;
            bf[j] = load_frag(p + boff, p + boff + 16);
        }
#pragma unroll
        for (int i = 0; i < 4; i++)
#pragma unroll
            for (int j = 0; j < 2; j++)
                acc[i][j] = wmma16(af[i], bf[j], acc[i][j]);
        __syncthreads();
    }
#endif

    // Epilogue. C layout: elem (vgpr r, lane) -> row = tile_m + r + 8*half, col = tile_n + l15
#pragma unroll
    for (int i = 0; i < 4; i++) {
        const int rbase = m0 + wm + i * 16 + 8 * half;
#pragma unroll
        for (int j = 0; j < 2; j++) {
            const int col = n0 + wn + j * 16 + l15;
            v8f c = acc[i][j];
            if constexpr (MODE == M_VT) {
                const int b = rbase >> 11, n = rbase & (NSEQ - 1);
                const int hh = col >> 6, d = col & 63;
                v8h hv;
#pragma unroll
                for (int r = 0; r < 8; r++) hv[r] = (_Float16)c[r];
                *(v8h*)(outH + ((size_t)((b * NH + hh) * DHD + d)) * NSEQ + n) = hv;
            } else {
#pragma unroll
                for (int r = 0; r < 8; r++) {
                    const int row = rbase + r;
                    float v = c[r];
                    if constexpr (MODE == M_QK) {
                        const int b = row >> 11, n = row & (NSEQ - 1);
                        const int hh = col >> 6, d = col & 63;
                        outH[(((size_t)(b * NH + hh) * NSEQ) + n) * DHD + d] = (_Float16)(v * qscale);
                    } else if constexpr (MODE == M_RES) {
                        const size_t idx = (size_t)row * Nn + col;
                        outF[idx] = v + res[idx];
                    } else if constexpr (MODE == M_GELU) {
                        v += bias[col];
                        float gel = 0.5f * v * (1.0f + erff(v * 0.70710678118f));
                        outH[(size_t)row * Nn + col] = (_Float16)gel;
                    } else { // M_FINAL
                        const size_t idx = (size_t)row * Nn + col;
                        outF[idx] = v + bias[col] + res[idx];
                    }
                }
            }
        }
    }
}

// ---------------- flash attention (causal), scale folded into Q ----------------
// Softmax denominator accumulated via WMMA against an all-ones B matrix:
// rowsum(P) = P x ones, replicated across lanes in the same C layout as O.
__global__ __launch_bounds__(256) void attn_kernel(
        const _Float16* __restrict__ Q, const _Float16* __restrict__ Km,
        const _Float16* __restrict__ Vt, _Float16* __restrict__ Aout) {
    __shared__ __align__(16) _Float16 P[8 * 16 * 40];   // per-wave 16x32 P tile, padded
    const int tid = threadIdx.x, lane = tid & 31, w = tid >> 5;
    const int l15 = lane & 15, half = lane >> 4;
    const int bh = blockIdx.x, b = bh >> 4, h = bh & 15;
    const int q0 = blockIdx.y * 128 + w * 16;

    const _Float16* Qb = Q  + (size_t)bh * NSEQ * DHD;
    const _Float16* Kb = Km + (size_t)bh * NSEQ * DHD;
    const _Float16* Vb = Vt + (size_t)bh * DHD * NSEQ;

    v16h aq[2];
    {
        const _Float16* qr = Qb + (size_t)(q0 + l15) * DHD + half * 8;
        aq[0] = load_frag(qr,      qr + 16);
        aq[1] = load_frag(qr + 32, qr + 48);
    }
    v16h ones;
#pragma unroll
    for (int i = 0; i < 16; i++) ones[i] = (_Float16)1.0f;

    v8f o[4];
#pragma unroll
    for (int t = 0; t < 4; t++) o[t] = vzero8();
    v8f ol = vzero8();          // softmax denominators (replicated per row)
    float mrun[8];
#pragma unroll
    for (int r = 0; r < 8; r++) mrun[r] = -1e30f;

    _Float16* Pw = P + w * 16 * 40;

    for (int j0 = 0; j0 <= q0 + 15; j0 += 32) {
        v8f s0 = vzero8(), s1 = vzero8();
#pragma unroll
        for (int c = 0; c < 2; c++) {
            const _Float16* k0p = Kb + (size_t)(j0 + l15) * DHD + c * 32 + half * 8;
            const _Float16* k1p = Kb + (size_t)(j0 + 16 + l15) * DHD + c * 32 + half * 8;
            v16h bk0 = load_frag(k0p, k0p + 16);
            v16h bk1 = load_frag(k1p, k1p + 16);
            s0 = wmma16(aq[c], bk0, s0);
            s1 = wmma16(aq[c], bk1, s1);
        }
#pragma unroll
        for (int r = 0; r < 8; r++) {
            const int row = q0 + r + 8 * half;
            float a0 = (j0 + l15      <= row) ? s0[r] : -1e30f;
            float a1 = (j0 + 16 + l15 <= row) ? s1[r] : -1e30f;
            float bm = fmaxf(a0, a1);
            bm = fmaxf(bm, __shfl_xor(bm, 1, 16));
            bm = fmaxf(bm, __shfl_xor(bm, 2, 16));
            bm = fmaxf(bm, __shfl_xor(bm, 4, 16));
            bm = fmaxf(bm, __shfl_xor(bm, 8, 16));
            const float mnew = fmaxf(mrun[r], bm);
            const float fac  = __expf(mrun[r] - mnew);
            const float p0   = __expf(a0 - mnew);
            const float p1   = __expf(a1 - mnew);
            mrun[r] = mnew;
            ol[r] = ol[r] * fac;
#pragma unroll
            for (int t = 0; t < 4; t++) o[t][r] = o[t][r] * fac;
            Pw[(r + 8 * half) * 40 + l15]      = (_Float16)p0;
            Pw[(r + 8 * half) * 40 + 16 + l15] = (_Float16)p1;
        }
        // re-fragment P (in-order LDS pipe within wave), then P*Vt^T and P*ones
        const _Float16* pp = Pw + l15 * 40 + half * 8;
        v16h ap = load_frag(pp, pp + 16);
        ol = wmma16(ap, ones, ol);
#pragma unroll
        for (int t = 0; t < 4; t++) {
            const _Float16* vp = Vb + (size_t)(t * 16 + l15) * NSEQ + j0 + half * 8;
            v16h bv = load_frag(vp, vp + 16);
            o[t] = wmma16(ap, bv, o[t]);
        }
    }

    float inv[8];
#pragma unroll
    for (int r = 0; r < 8; r++) inv[r] = 1.0f / ol[r];
#pragma unroll
    for (int t = 0; t < 4; t++) {
        const int col = h * DHD + t * 16 + l15;
#pragma unroll
        for (int r = 0; r < 8; r++) {
            const int n = q0 + r + 8 * half;
            Aout[((size_t)(b * NSEQ + n)) * ED + col] = (_Float16)(o[t][r] * inv[r]);
        }
    }
}

// ---------------- host launcher ----------------
extern "C" void kernel_launch(void* const* d_in, const int* in_sizes, int n_in,
                              void* d_out, int out_size, void* d_ws, size_t ws_size,
                              hipStream_t stream) {
    const float* x     = (const float*)d_in[0];
    // d_in[1] = mask (causal tril) — structure hardcoded
    const float* W_Q   = (const float*)d_in[2];
    const float* W_K   = (const float*)d_in[3];
    const float* W_V   = (const float*)d_in[4];
    const float* W_O   = (const float*)d_in[5];
    const float* ln1_g = (const float*)d_in[6];
    const float* ln1_b = (const float*)d_in[7];
    const float* ln2_g = (const float*)d_in[8];
    const float* ln2_b = (const float*)d_in[9];
    const float* fc1_w = (const float*)d_in[10];
    const float* fc1_b = (const float*)d_in[11];
    const float* fc2_w = (const float*)d_in[12];
    const float* fc2_b = (const float*)d_in[13];
    float* out = (float*)d_out;

    char* ws = (char*)d_ws;
    size_t off = 0;
    auto alloc = [&](size_t bytes) -> void* {
        void* p = ws + off;
        off += (bytes + 255) & ~(size_t)255;
        return p;
    };
    _Float16* wq16   = (_Float16*)alloc((size_t)ED * ED * 2);
    _Float16* wk16   = (_Float16*)alloc((size_t)ED * ED * 2);
    _Float16* wv16   = (_Float16*)alloc((size_t)ED * ED * 2);
    _Float16* wo16   = (_Float16*)alloc((size_t)ED * ED * 2);
    _Float16* fc1w16 = (_Float16*)alloc((size_t)HD * ED * 2);
    _Float16* fc2w16 = (_Float16*)alloc((size_t)ED * HD * 2);
    _Float16* hbuf   = (_Float16*)alloc((size_t)NTOK * ED * 2);
    _Float16* Qbuf   = (_Float16*)alloc((size_t)NTOK * ED * 2);
    _Float16* Kbuf   = (_Float16*)alloc((size_t)NTOK * ED * 2);
    _Float16* Vtbuf  = (_Float16*)alloc((size_t)NTOK * ED * 2);
    _Float16* attno  = (_Float16*)alloc((size_t)NTOK * ED * 2);
    float*    x1     = (float*)   alloc((size_t)NTOK * ED * 4);
    _Float16* h2     = (_Float16*)alloc((size_t)NTOK * ED * 2);
    _Float16* ffn1   = (_Float16*)alloc((size_t)NTOK * HD * 2);

    auto cvt = [&](const float* src, _Float16* dst, int n) {
        int n4 = n / 4;
        cvt_kernel<<<(n4 + 255) / 256, 256, 0, stream>>>(src, dst, n4);
    };
    cvt(W_Q, wq16, ED * ED);
    cvt(W_K, wk16, ED * ED);
    cvt(W_V, wv16, ED * ED);
    cvt(W_O, wo16, ED * ED);
    cvt(fc1_w, fc1w16, HD * ED);
    cvt(fc2_w, fc2w16, ED * HD);

    // LN1
    ln_kernel<<<NTOK, 256, 0, stream>>>(x, ln1_g, ln1_b, hbuf);

    dim3 gP(ED / 128, NTOK / 128);   // (8, 32) for N=1024 outputs
    dim3 gF(HD / 128, NTOK / 128);   // (32, 32) for N=4096 outputs

    // Q (scale folded), K, V^T
    gemm_kernel<M_QK><<<gP, 256, 0, stream>>>(hbuf, wq16, nullptr, nullptr, nullptr, Qbuf, ED, ED, 0.125f);
    gemm_kernel<M_QK><<<gP, 256, 0, stream>>>(hbuf, wk16, nullptr, nullptr, nullptr, Kbuf, ED, ED, 1.0f);
    gemm_kernel<M_VT><<<gP, 256, 0, stream>>>(hbuf, wv16, nullptr, nullptr, nullptr, Vtbuf, ED, ED, 1.0f);

    // flash attention
    attn_kernel<<<dim3(NB * NH, NSEQ / 128), 256, 0, stream>>>(Qbuf, Kbuf, Vtbuf, attno);

    // out projection + residual (no bias on W_O)
    gemm_kernel<M_RES><<<gP, 256, 0, stream>>>(attno, wo16, nullptr, x, x1, nullptr, ED, ED, 1.0f);

    // LN2
    ln_kernel<<<NTOK, 256, 0, stream>>>(x1, ln2_g, ln2_b, h2);

    // FFN
    gemm_kernel<M_GELU><<<gF, 256, 0, stream>>>(h2, fc1w16, fc1_b, nullptr, nullptr, ffn1, HD, ED, 1.0f);
    gemm_kernel<M_FINAL><<<gP, 256, 0, stream>>>(ffn1, fc2w16, fc2_b, x1, out, nullptr, ED, HD, 1.0f);

    (void)in_sizes; (void)n_in; (void)out_size; (void)ws_size;
}
